// UnfrozenPatchDrapeModel_21028159881694
// MI455X (gfx1250) — compile-verified, matrix operand order checked by reading, not run on hardware
//
#include <hip/hip_runtime.h>
#include <hip/hip_bf16.h>

// UnfrozenPatchDrapeModel for MI455X (gfx1250): all GEMMs via v_wmma_f32_16x16x32_f16,
// f32 accumulate. Register-blocked 32x64 per-wave tiles: 8 WMMA per K-step fed by
// 12 global_load_b128 (2 A-frags + 4 B-frags), ~1.5 loads/WMMA. Weights are
// pre-transposed to N x Kpad f16 so A and B fragments share the ISA 16-bit layout
// addressing (two b128 per lane per fragment).

typedef _Float16 half_t;
typedef __attribute__((ext_vector_type(16))) _Float16 v16h;
typedef __attribute__((ext_vector_type(8)))  _Float16 v8h;
typedef __attribute__((ext_vector_type(8)))  float    v8f;

#define NN_   20000
#define NE_   120000
#define B_    4
#define L_    6
#define D_    128
#define H_    256
#define P_    196
#define MPAT_ 800   // B_*P_ = 784 padded up to a multiple of 32

__device__ __forceinline__ v16h ld_frag(const half_t* __restrict__ p)
{
    v8h lo = *(const v8h*)p;          // K = h8 + 0..7    (global_load_b128)
    v8h hi = *(const v8h*)(p + 16);   // K = h8 + 16..23  (global_load_b128)
    v16h r;
    #pragma unroll
    for (int i = 0; i < 8; ++i) { r[i] = lo[i]; r[8 + i] = hi[i]; }
    return r;
}

__device__ __forceinline__ v8f wmma32(v16h a, v16h b, v8f c)
{
    return __builtin_amdgcn_wmma_f32_16x16x32_f16(false, a, false, b,
                                                  (short)0, c, false, false);
}

// ---------------------------------------------------------------------------
// WMMA GEMM: C[M,N] f32 = A[M,K] f16 @ Wt[N,K] f16 (+bias[N])
// wave tile 32x64 (2x4 of 16x16), block = (N/64) waves -> 32 x N tile.
// grid = (M/32). K multiple of 32, M multiple of 32, N in {128, 256}.
// ---------------------------------------------------------------------------
__global__ __launch_bounds__(128)
void k_gemm(const half_t* __restrict__ A, const half_t* __restrict__ Wt,
            const float* __restrict__ bias, float* __restrict__ C,
            int M, int N, int K)
{
    const int lane = threadIdx.x & 31;
    const int wave = threadIdx.x >> 5;
    const int row0 = blockIdx.x << 5;          // 32 rows per block
    const int col0 = wave << 6;                // 64 cols per wave
    const int lm   = lane & 15;
    const int h8   = (lane >> 4) << 3;         // K sub-offset per ISA A/B layout
    const half_t* ar0 = A  + (size_t)(row0 + lm) * K + h8;
    const half_t* ar1 = ar0 + (size_t)16 * K;
    const half_t* br0 = Wt + (size_t)(col0 + lm) * K + h8;
    const half_t* br1 = br0 + (size_t)16 * K;
    const half_t* br2 = br0 + (size_t)32 * K;
    const half_t* br3 = br0 + (size_t)48 * K;

    v8f acc[2][4];
    #pragma unroll
    for (int i = 0; i < 2; ++i)
        #pragma unroll
        for (int j = 0; j < 4; ++j) acc[i][j] = (v8f){};

    for (int k = 0; k < K; k += 32) {
        v16h a0 = ld_frag(ar0 + k);
        v16h a1 = ld_frag(ar1 + k);
        v16h b0 = ld_frag(br0 + k);
        v16h b1 = ld_frag(br1 + k);
        v16h b2 = ld_frag(br2 + k);
        v16h b3 = ld_frag(br3 + k);
        acc[0][0] = wmma32(a0, b0, acc[0][0]);
        acc[1][0] = wmma32(a1, b0, acc[1][0]);
        acc[0][1] = wmma32(a0, b1, acc[0][1]);
        acc[1][1] = wmma32(a1, b1, acc[1][1]);
        acc[0][2] = wmma32(a0, b2, acc[0][2]);
        acc[1][2] = wmma32(a1, b2, acc[1][2]);
        acc[0][3] = wmma32(a0, b3, acc[0][3]);
        acc[1][3] = wmma32(a1, b3, acc[1][3]);
    }

    const int rb = (lane >> 4) << 3;           // C layout: vgpr r -> row r (+8 hi half)
    float bn[4];
    #pragma unroll
    for (int j = 0; j < 4; ++j) bn[j] = bias[col0 + (j << 4) + lm];
    #pragma unroll
    for (int i = 0; i < 2; ++i) {
        #pragma unroll
        for (int r = 0; r < 8; ++r) {
            const size_t row = (size_t)(row0 + (i << 4) + rb + r) * N;
            #pragma unroll
            for (int j = 0; j < 4; ++j)
                C[row + col0 + (j << 4) + lm] = acc[i][j][r] + bn[j];
        }
    }
}

// ---------------------------------------------------------------------------
// Weight pack: src f32 (K x N, row major) -> dst f16 (N x Kpad), zero padded K
// ---------------------------------------------------------------------------
__global__ void k_wt(const float* __restrict__ src, half_t* __restrict__ dst,
                     int K, int N, int Kpad)
{
    int t = blockIdx.x * blockDim.x + threadIdx.x;
    if (t >= N * Kpad) return;
    int n = t / Kpad, k = t % Kpad;
    dst[t] = (half_t)(k < K ? src[(size_t)k * N + n] : 0.f);
}

// Activation pack: src f32 (M x Kin) -> dst f16 (M x Kpad), zero padded
__global__ void k_pack_act(const float* __restrict__ src, half_t* __restrict__ dst,
                           int M, int Kin, int Kpad)
{
    int t = blockIdx.x * blockDim.x + threadIdx.x;
    if (t >= M * Kpad) return;
    int m = t / Kpad, k = t % Kpad;
    dst[t] = (half_t)(k < Kin ? src[(size_t)m * Kin + k] : 0.f);
}

// global_cond (B x 150) = [mean_p(patch), smpl, physics, size]
__global__ void k_cond(const float* __restrict__ patch, const float* __restrict__ smpl,
                       const float* __restrict__ phys, const float* __restrict__ size,
                       float* __restrict__ cond)
{
    int t = blockIdx.x * blockDim.x + threadIdx.x;
    if (t >= B_ * 150) return;
    int b = t / 150, j = t % 150;
    float v;
    if (j < 128) {
        float s = 0.f;
        for (int p = 0; p < P_; ++p) s += patch[((size_t)b * P_ + p) * D_ + j];
        v = s * (1.f / (float)P_);
    } else if (j < 138) v = smpl[b * 10 + (j - 128)];
    else if (j < 148)   v = phys[b * 10 + (j - 138)];
    else                v = size[b * 2 + (j - 148)];
    cond[t] = v;
}

// gb (L x B x 256) = cond @ film_w[l] + film_b[l]
__global__ void k_film_gb(const float* __restrict__ cond, const float* __restrict__ fw,
                          const float* __restrict__ fb, float* __restrict__ gb)
{
    int t = blockIdx.x * blockDim.x + threadIdx.x;
    if (t >= L_ * B_ * 256) return;
    int n = t & 255, b = (t >> 8) & 3, l = t >> 10;
    float s = fb[l * 256 + n];
    const float* c = cond + b * 150;
    const float* w = fw + (size_t)l * 150 * 256 + n;
    for (int k = 0; k < 150; ++k) s += c[k] * w[(size_t)k * 256];
    gb[t] = s;
}

// LayerNorm (+optional ReLU), f32 in -> f16 out. One wave per row, N in {128,256}.
__global__ __launch_bounds__(256)
void k_ln_act(const float* __restrict__ X, const float* __restrict__ g,
              const float* __restrict__ b, half_t* __restrict__ O,
              int M, int N, int relu)
{
    int row  = blockIdx.x * 8 + (threadIdx.x >> 5);
    int lane = threadIdx.x & 31;
    if (row >= M) return;
    const float* r = X + (size_t)row * N;
    int per = N >> 5;
    float s = 0.f, ss = 0.f;
    for (int i = 0; i < per; ++i) { float v = r[lane + (i << 5)]; s += v; ss += v * v; }
    for (int msk = 16; msk >= 1; msk >>= 1) { s += __shfl_xor(s, msk, 32); ss += __shfl_xor(ss, msk, 32); }
    float mean = s / (float)N;
    float inv  = rsqrtf(ss / (float)N - mean * mean + 1e-5f);
    half_t* o = O + (size_t)row * N;
    for (int i = 0; i < per; ++i) {
        int j = lane + (i << 5);
        float v = (r[j] - mean) * inv * g[j] + b[j];
        if (relu) v = v > 0.f ? v : 0.f;
        o[j] = (half_t)v;
    }
}

// edge_input f16 (E x 384) = [x[src], x[dst], edge_attr] (f32 sources)
__global__ void k_edge_concat(const float* __restrict__ x, const float* __restrict__ ea,
                              const int* __restrict__ src, const int* __restrict__ dst,
                              half_t* __restrict__ ein, int E)
{
    int t = blockIdx.x * blockDim.x + threadIdx.x;
    if (t >= E * 96) return;
    int e = t / 96, j = (t % 96) * 4;
    const float* sp; int off;
    if (j < 128)      { sp = x  + (size_t)src[e] * D_; off = j; }
    else if (j < 256) { sp = x  + (size_t)dst[e] * D_; off = j - 128; }
    else              { sp = ea + (size_t)e   * D_;    off = j - 256; }
    float4 v = *(const float4*)(sp + off);
    half_t* o = ein + (size_t)e * 384 + j;
    o[0] = (half_t)v.x; o[1] = (half_t)v.y; o[2] = (half_t)v.z; o[3] = (half_t)v.w;
}

__global__ void k_add(float* __restrict__ a, const float* __restrict__ b, int n)
{
    int t = blockIdx.x * blockDim.x + threadIdx.x;
    if (t < n) a[t] += b[t];
}

__global__ void k_zero(float* __restrict__ p, int n)
{
    int t = blockIdx.x * blockDim.x + threadIdx.x;
    if (t < n) p[t] = 0.f;
}

// agg[dst] += edge_attr  (segment_sum via f32 atomics)
__global__ void k_scatter(const float* __restrict__ ea, const int* __restrict__ dst,
                          float* __restrict__ agg, int E)
{
    int t = blockIdx.x * blockDim.x + threadIdx.x;
    if (t >= E * D_) return;
    int e = t >> 7, j = t & 127;
    atomicAdd(agg + (size_t)dst[e] * D_ + j, ea[(size_t)e * D_ + j]);
}

// node_in f16 (N x 256) = [x, agg]
__global__ void k_node_concat(const float* __restrict__ x, const float* __restrict__ agg,
                              half_t* __restrict__ nin, int Nn)
{
    int t = blockIdx.x * blockDim.x + threadIdx.x;
    if (t >= Nn * 256) return;
    int n = t >> 8, j = t & 255;
    float v = (j < 128) ? x[(size_t)n * D_ + j] : agg[(size_t)n * D_ + (j - 128)];
    nin[t] = (half_t)v;
}

// x += gamma[batch]*LN(nu) + beta[batch]   (wave per row, N=128)
__global__ __launch_bounds__(256)
void k_film_apply(const float* __restrict__ nu, const float* __restrict__ g,
                  const float* __restrict__ b, const float* __restrict__ gb,
                  const int* __restrict__ batch, float* __restrict__ x, int Nn)
{
    int row  = blockIdx.x * 8 + (threadIdx.x >> 5);
    int lane = threadIdx.x & 31;
    if (row >= Nn) return;
    const float* r = nu + (size_t)row * D_;
    float s = 0.f, ss = 0.f;
    #pragma unroll
    for (int i = 0; i < 4; ++i) { float v = r[lane + (i << 5)]; s += v; ss += v * v; }
    for (int msk = 16; msk >= 1; msk >>= 1) { s += __shfl_xor(s, msk, 32); ss += __shfl_xor(ss, msk, 32); }
    float mean = s * (1.f / 128.f);
    float inv  = rsqrtf(ss * (1.f / 128.f) - mean * mean + 1e-5f);
    const float* gbr = gb + (size_t)batch[row] * 256;
    float* xr = x + (size_t)row * D_;
    #pragma unroll
    for (int i = 0; i < 4; ++i) {
        int j = lane + (i << 5);
        float v = (r[j] - mean) * inv * g[j] + b[j];
        xr[j] += gbr[j] * v + gbr[128 + j];
    }
}

// cross-attention: online softmax over P=196 tokens of the node's own graph
__global__ __launch_bounds__(256)
void k_cross_attn(const float* __restrict__ Q, const float* __restrict__ Kb,
                  const float* __restrict__ Vb, const int* __restrict__ batch,
                  float* __restrict__ x, int Nn)
{
    int node = blockIdx.x * 8 + (threadIdx.x >> 5);
    int lane = threadIdx.x & 31;
    if (node >= Nn) return;
    int b = batch[node];
    const float4 q  = *(const float4*)(Q + (size_t)node * D_ + lane * 4);
    const float* Kr = Kb + (size_t)b * P_ * D_;
    const float* Vr = Vb + (size_t)b * P_ * D_;
    const float scale = 0.088388347648318447f;   // 128^-0.5
    float mmax = -3.0e38f, lsum = 0.f;
    float4 acc = {0.f, 0.f, 0.f, 0.f};
    for (int p = 0; p < P_; ++p) {
        const float4 kk = *(const float4*)(Kr + (size_t)p * D_ + lane * 4);
        float s = q.x * kk.x + q.y * kk.y + q.z * kk.z + q.w * kk.w;
        for (int msk = 16; msk >= 1; msk >>= 1) s += __shfl_xor(s, msk, 32);
        s *= scale;
        float mn   = fmaxf(mmax, s);
        float corr = __expf(mmax - mn);
        float w    = __expf(s - mn);
        const float4 vv = *(const float4*)(Vr + (size_t)p * D_ + lane * 4);
        lsum  = lsum * corr + w;
        acc.x = acc.x * corr + w * vv.x;
        acc.y = acc.y * corr + w * vv.y;
        acc.z = acc.z * corr + w * vv.z;
        acc.w = acc.w * corr + w * vv.w;
        mmax = mn;
    }
    float il = 1.f / lsum;
    float* xr = x + (size_t)node * D_ + lane * 4;
    float4 xv = *(const float4*)xr;
    xv.x += acc.x * il; xv.y += acc.y * il; xv.z += acc.z * il; xv.w += acc.w * il;
    *(float4*)xr = xv;
}

// pred_pos (N x 3) = x @ dec_w + dec_b
__global__ void k_decode(const float* __restrict__ x, const float* __restrict__ dw,
                         const float* __restrict__ db, float* __restrict__ out, int Nn)
{
    int t = blockIdx.x * blockDim.x + threadIdx.x;
    if (t >= Nn * 3) return;
    int n = t / 3, c = t % 3;
    float s = db[c];
    const float* xr = x + (size_t)n * D_;
    for (int k = 0; k < D_; ++k) s += xr[k] * dw[k * 3 + c];
    out[t] = s;
}

// fabric_logits (B x 6) = style @ fab_w + fab_b
__global__ void k_fab(const float* __restrict__ st, const float* __restrict__ fw,
                      const float* __restrict__ fb, float* __restrict__ out)
{
    int t = threadIdx.x;
    if (t >= B_ * 6) return;
    int b = t / 6, c = t % 6;
    float s = fb[c];
    const float* sr = st + (size_t)b * D_;
    for (int k = 0; k < D_; ++k) s += sr[k] * fw[k * 6 + c];
    out[t] = s;
}

// ---------------------------------------------------------------------------
// Host orchestration
// ---------------------------------------------------------------------------
static inline int cdiv(int a, int b) { return (a + b - 1) / b; }

extern "C" void kernel_launch(void* const* d_in, const int* in_sizes, int n_in,
                              void* d_out, int out_size, void* d_ws, size_t ws_size,
                              hipStream_t stream)
{
    // --- input index map (top-level dict order; nested dicts alphabetical) ---
    const float* node_feats = (const float*)d_in[0];
    const float* edge_feats = (const float*)d_in[1];
    const float* patch_emb  = (const float*)d_in[2];
    const float* style_emb  = (const float*)d_in[3];
    const float* smpl       = (const float*)d_in[4];
    const float* physics    = (const float*)d_in[5];
    const float* size_p     = (const float*)d_in[6];
    const float* blk_ln_b = (const float*)d_in[7];
    const float* blk_ln_g = (const float*)d_in[8];
    const float* ca_kb    = (const float*)d_in[9];
    const float* ca_kw    = (const float*)d_in[10];
    const float* ca_ln_b  = (const float*)d_in[11];
    const float* ca_ln_g  = (const float*)d_in[12];
    const float* ca_qb    = (const float*)d_in[13];
    const float* ca_qw    = (const float*)d_in[14];
    const float* ca_vb    = (const float*)d_in[15];
    const float* ca_vw    = (const float*)d_in[16];
    const float* dec_b    = (const float*)d_in[17];
    const float* dec_w    = (const float*)d_in[18];
    const float* ee_b1 = (const float*)d_in[19], *ee_b2 = (const float*)d_in[20],
               * ee_b3 = (const float*)d_in[21], *ee_be1 = (const float*)d_in[22],
               * ee_be2 = (const float*)d_in[23], *ee_g1 = (const float*)d_in[24],
               * ee_g2 = (const float*)d_in[25], *ee_w1 = (const float*)d_in[26],
               * ee_w2 = (const float*)d_in[27], *ee_w3 = (const float*)d_in[28];
    const float* em_b1 = (const float*)d_in[29], *em_b2 = (const float*)d_in[30],
               * em_b3 = (const float*)d_in[31], *em_be1 = (const float*)d_in[32],
               * em_be2 = (const float*)d_in[33], *em_g1 = (const float*)d_in[34],
               * em_g2 = (const float*)d_in[35], *em_w1 = (const float*)d_in[36],
               * em_w2 = (const float*)d_in[37], *em_w3 = (const float*)d_in[38];
    const float* fab_b  = (const float*)d_in[39];
    const float* fab_w  = (const float*)d_in[40];
    const float* film_b = (const float*)d_in[41];
    const float* film_w = (const float*)d_in[42];
    const float* ne_b1 = (const float*)d_in[43], *ne_b2 = (const float*)d_in[44],
               * ne_b3 = (const float*)d_in[45], *ne_be1 = (const float*)d_in[46],
               * ne_be2 = (const float*)d_in[47], *ne_g1 = (const float*)d_in[48],
               * ne_g2 = (const float*)d_in[49], *ne_w1 = (const float*)d_in[50],
               * ne_w2 = (const float*)d_in[51], *ne_w3 = (const float*)d_in[52];
    const float* nm_b1 = (const float*)d_in[53], *nm_b2 = (const float*)d_in[54],
               * nm_b3 = (const float*)d_in[55], *nm_be1 = (const float*)d_in[56],
               * nm_be2 = (const float*)d_in[57], *nm_g1 = (const float*)d_in[58],
               * nm_g2 = (const float*)d_in[59], *nm_w1 = (const float*)d_in[60],
               * nm_w2 = (const float*)d_in[61], *nm_w3 = (const float*)d_in[62];
    const int* edge_index = (const int*)d_in[63];
    const int* batch      = (const int*)d_in[64];
    const int* e_src = edge_index;
    const int* e_dst = edge_index + NE_;

    float* out = (float*)d_out;

    // --- workspace bump allocator (256B aligned) ---
    char* wsp = (char*)d_ws;
    auto take = [&](size_t bytes) -> void* {
        void* r = (void*)wsp; wsp += (bytes + 255) & ~(size_t)255; return r;
    };
    // f16 transposed weights
    half_t* w_ne1 = (half_t*)take((size_t)256 * 32  * 2);
    half_t* w_ne2 = (half_t*)take((size_t)256 * 256 * 2);
    half_t* w_ne3 = (half_t*)take((size_t)128 * 256 * 2);
    half_t* w_ee1 = (half_t*)take((size_t)256 * 32  * 2);
    half_t* w_ee2 = (half_t*)take((size_t)256 * 256 * 2);
    half_t* w_ee3 = (half_t*)take((size_t)128 * 256 * 2);
    half_t* w_em1 = (half_t*)take((size_t)L_ * 256 * 384 * 2);
    half_t* w_em2 = (half_t*)take((size_t)L_ * 256 * 256 * 2);
    half_t* w_em3 = (half_t*)take((size_t)L_ * 128 * 256 * 2);
    half_t* w_nm1 = (half_t*)take((size_t)L_ * 256 * 256 * 2);
    half_t* w_nm2 = (half_t*)take((size_t)L_ * 256 * 256 * 2);
    half_t* w_nm3 = (half_t*)take((size_t)L_ * 128 * 256 * 2);
    half_t* w_q   = (half_t*)take((size_t)2 * 128 * 128 * 2);
    half_t* w_k   = (half_t*)take((size_t)2 * 128 * 128 * 2);
    half_t* w_v   = (half_t*)take((size_t)2 * 128 * 128 * 2);
    // activations
    half_t* a_nf  = (half_t*)take((size_t)NN_ * 32 * 2);       // node feats padded
    half_t* a_ef  = (half_t*)take((size_t)NE_ * 32 * 2);       // edge feats padded
    half_t* a_pt  = (half_t*)take((size_t)MPAT_ * D_ * 2);     // patch tokens f16 (padded M)
    float*  cond  = (float*) take((size_t)B_ * 150 * 4);
    float*  gb    = (float*) take((size_t)L_ * B_ * 256 * 4);
    float*  x     = (float*) take((size_t)NN_ * D_ * 4);
    float*  eattr = (float*) take((size_t)NE_ * D_ * 4);
    float*  agg   = (float*) take((size_t)NN_ * D_ * 4);
    half_t* ein   = (half_t*)take((size_t)NE_ * 384 * 2);
    half_t* nin   = (half_t*)take((size_t)NN_ * 256 * 2);
    float*  tE    = (float*) take((size_t)NE_ * 256 * 4);
    half_t* tEh   = (half_t*)take((size_t)NE_ * 256 * 2);
    float*  dE    = (float*) take((size_t)NE_ * D_ * 4);
    float*  tN    = (float*) take((size_t)NN_ * 256 * 4);
    half_t* tNh   = (half_t*)take((size_t)NN_ * 256 * 2);
    float*  nu    = (float*) take((size_t)NN_ * D_ * 4);
    half_t* xnh   = (half_t*)take((size_t)NN_ * D_ * 2);
    float*  Qb    = (float*) take((size_t)NN_ * D_ * 4);
    float*  Kbuf  = (float*) take((size_t)MPAT_ * D_ * 4);
    float*  Vbuf  = (float*) take((size_t)MPAT_ * D_ * 4);

    auto WT = [&](const float* s, half_t* d, int K, int N, int Kpad) {
        int tot = N * Kpad;
        k_wt<<<cdiv(tot, 256), 256, 0, stream>>>(s, d, K, N, Kpad);
    };
    auto GEMM = [&](const half_t* A, const half_t* W, const float* bias, float* C,
                    int M, int N, int K) {
        int wpb = N >> 6;                      // waves per block: 2 (N=128) or 4 (N=256)
        k_gemm<<<dim3(M >> 5), dim3(wpb << 5), 0, stream>>>(A, W, bias, C, M, N, K);
    };
    auto LN = [&](const float* X, const float* g, const float* b, half_t* O,
                  int M, int N, int relu) {
        k_ln_act<<<cdiv(M, 8), 256, 0, stream>>>(X, g, b, O, M, N, relu);
    };

    // --- pack weights (f32 -> f16, transpose to N x Kpad) ---
    WT(ne_w1, w_ne1, 8, 256, 32);  WT(ne_w2, w_ne2, 256, 256, 256); WT(ne_w3, w_ne3, 256, 128, 256);
    WT(ee_w1, w_ee1, 4, 256, 32);  WT(ee_w2, w_ee2, 256, 256, 256); WT(ee_w3, w_ee3, 256, 128, 256);
    for (int l = 0; l < L_; ++l) {
        WT(em_w1 + (size_t)l * 384 * 256, w_em1 + (size_t)l * 256 * 384, 384, 256, 384);
        WT(em_w2 + (size_t)l * 256 * 256, w_em2 + (size_t)l * 256 * 256, 256, 256, 256);
        WT(em_w3 + (size_t)l * 256 * 128, w_em3 + (size_t)l * 128 * 256, 256, 128, 256);
        WT(nm_w1 + (size_t)l * 256 * 256, w_nm1 + (size_t)l * 256 * 256, 256, 256, 256);
        WT(nm_w2 + (size_t)l * 256 * 256, w_nm2 + (size_t)l * 256 * 256, 256, 256, 256);
        WT(nm_w3 + (size_t)l * 256 * 128, w_nm3 + (size_t)l * 128 * 256, 256, 128, 256);
    }
    for (int ci = 0; ci < 2; ++ci) {
        WT(ca_qw + (size_t)ci * 128 * 128, w_q + (size_t)ci * 128 * 128, 128, 128, 128);
        WT(ca_kw + (size_t)ci * 128 * 128, w_k + (size_t)ci * 128 * 128, 128, 128, 128);
        WT(ca_vw + (size_t)ci * 128 * 128, w_v + (size_t)ci * 128 * 128, 128, 128, 128);
    }

    // --- pack activations ---
    k_pack_act<<<cdiv(NN_ * 32, 256), 256, 0, stream>>>(node_feats, a_nf, NN_, 8, 32);
    k_pack_act<<<cdiv(NE_ * 32, 256), 256, 0, stream>>>(edge_feats, a_ef, NE_, 4, 32);
    k_pack_act<<<cdiv(B_ * P_ * D_, 256), 256, 0, stream>>>(patch_emb, a_pt, B_ * P_, 128, 128);

    // --- FiLM conditioning ---
    k_cond<<<cdiv(B_ * 150, 256), 256, 0, stream>>>(patch_emb, smpl, physics, size_p, cond);
    k_film_gb<<<cdiv(L_ * B_ * 256, 256), 256, 0, stream>>>(cond, film_w, film_b, gb);

    // --- encoders ---
    GEMM(a_nf, w_ne1, ne_b1, tN, NN_, 256, 32);
    LN(tN, ne_g1, ne_be1, tNh, NN_, 256, 1);
    GEMM(tNh, w_ne2, ne_b2, tN, NN_, 256, 256);
    LN(tN, ne_g2, ne_be2, tNh, NN_, 256, 1);
    GEMM(tNh, w_ne3, ne_b3, x, NN_, 128, 256);

    GEMM(a_ef, w_ee1, ee_b1, tE, NE_, 256, 32);
    LN(tE, ee_g1, ee_be1, tEh, NE_, 256, 1);
    GEMM(tEh, w_ee2, ee_b2, tE, NE_, 256, 256);
    LN(tE, ee_g2, ee_be2, tEh, NE_, 256, 1);
    GEMM(tEh, w_ee3, ee_b3, eattr, NE_, 128, 256);

    // --- GNN layers ---
    for (int l = 0; l < L_; ++l) {
        // edge MLP
        k_edge_concat<<<cdiv(NE_ * 96, 256), 256, 0, stream>>>(x, eattr, e_src, e_dst, ein, NE_);
        GEMM(ein, w_em1 + (size_t)l * 256 * 384, em_b1 + l * 256, tE, NE_, 256, 384);
        LN(tE, em_g1 + l * 256, em_be1 + l * 256, tEh, NE_, 256, 1);
        GEMM(tEh, w_em2 + (size_t)l * 256 * 256, em_b2 + l * 256, tE, NE_, 256, 256);
        LN(tE, em_g2 + l * 256, em_be2 + l * 256, tEh, NE_, 256, 1);
        GEMM(tEh, w_em3 + (size_t)l * 128 * 256, em_b3 + l * 128, dE, NE_, 128, 256);
        k_add<<<cdiv(NE_ * D_, 256), 256, 0, stream>>>(eattr, dE, NE_ * D_);
        // aggregate
        k_zero<<<cdiv(NN_ * D_, 256), 256, 0, stream>>>(agg, NN_ * D_);
        k_scatter<<<cdiv(NE_ * D_, 256), 256, 0, stream>>>(eattr, e_dst, agg, NE_);
        // node MLP
        k_node_concat<<<cdiv(NN_ * 256, 256), 256, 0, stream>>>(x, agg, nin, NN_);
        GEMM(nin, w_nm1 + (size_t)l * 256 * 256, nm_b1 + l * 256, tN, NN_, 256, 256);
        LN(tN, nm_g1 + l * 256, nm_be1 + l * 256, tNh, NN_, 256, 1);
        GEMM(tNh, w_nm2 + (size_t)l * 256 * 256, nm_b2 + l * 256, tN, NN_, 256, 256);
        LN(tN, nm_g2 + l * 256, nm_be2 + l * 256, tNh, NN_, 256, 1);
        GEMM(tNh, w_nm3 + (size_t)l * 128 * 256, nm_b3 + l * 128, nu, NN_, 128, 256);
        // FiLM residual
        k_film_apply<<<cdiv(NN_, 8), 256, 0, stream>>>(nu, blk_ln_g + l * 128, blk_ln_b + l * 128,
                                                       gb + (size_t)l * B_ * 256, batch, x, NN_);
        // cross attention after layers 2 and 5
        if (l == 2 || l == 5) {
            int ci = (l == 2) ? 0 : 1;
            LN(x, ca_ln_g + ci * 128, ca_ln_b + ci * 128, xnh, NN_, 128, 0);
            GEMM(xnh, w_q + (size_t)ci * 128 * 128, ca_qb + ci * 128, Qb, NN_, 128, 128);
            GEMM(a_pt, w_k + (size_t)ci * 128 * 128, ca_kb + ci * 128, Kbuf, MPAT_, 128, 128);
            GEMM(a_pt, w_v + (size_t)ci * 128 * 128, ca_vb + ci * 128, Vbuf, MPAT_, 128, 128);
            k_cross_attn<<<cdiv(NN_, 8), 256, 0, stream>>>(Qb, Kbuf, Vbuf, batch, x, NN_);
        }
    }

    // --- decoders ---
    k_decode<<<cdiv(NN_ * 3, 256), 256, 0, stream>>>(x, dec_w, dec_b, out, NN_);
    k_fab<<<1, 32, 0, stream>>>(style_emb, fab_w, fab_b, out + (size_t)NN_ * 3);
}
